// MultiHeadAttention_4260607558121
// MI455X (gfx1250) — compile-verified, matrix operand order checked by reading, not run on hardware
//
#include <hip/hip_runtime.h>

// MHA block for MI455X (gfx1250, wave32, WMMA 16x16x32 f16->f32 accumulate).
// Pipeline:
//   prep:    fold BN + bias + 1/sqrt(D) into f16 weights
//   proj x3: QP/KP -> [b,h,L,32] f16 (score scale folded into Q), VP -> [b,h,32,L] f16
//   attn:    flash-style online softmax, 16 queries/wave, 64 keys/iter, P transposed via LDS
//   outproj: W_o GEMM + bias + residual, fp32 out

typedef __attribute__((ext_vector_type(16))) _Float16 v16h;
typedef __attribute__((ext_vector_type(8)))  _Float16 v8h;
typedef __attribute__((ext_vector_type(8)))  float    v8f;

#define WMMA_F16(a, b, c) \
  __builtin_amdgcn_wmma_f32_16x16x32_f16(false, (a), false, (b), (short)0, (c), false, false)

static constexpr int NB   = 4;     // batch
static constexpr int CC   = 256;   // channels (CQ == CVK)
static constexpr int LSEQ = 2048;  // sequence length
static constexpr int NH   = 8;     // heads
static constexpr int DD   = 32;    // depth per head

// ---- workspace layout (bytes) ----
static constexpr size_t WS_WQH  = 0;               // 256*256 f16
static constexpr size_t WS_WKH  = 131072;
static constexpr size_t WS_WVH  = 262144;
static constexpr size_t WS_WOH  = 393216;
static constexpr size_t WS_BQ   = 524288;          // 256 f32 each
static constexpr size_t WS_BK   = 525312;
static constexpr size_t WS_BV   = 526336;
static constexpr size_t WS_QP   = 528384;          // B*H*L*D f16 = 4 MiB
static constexpr size_t WS_KP   = WS_QP + 4194304;
static constexpr size_t WS_VP   = WS_KP + 4194304;
static constexpr size_t WS_ATT  = WS_VP + 4194304; // [B*L, 256] f16

// Load a 16x32 A/B fragment (f16) from a row-major matrix, tile origin (row0,k0).
// Per-lane: row = row0 + (lane&15); elems 0..7 = K kb..kb+7, 8..15 = kb+16..kb+23,
// kb = k0 + 8*(lane>=16).  Matches CDNA5 16-bit A-matrix VGPR layout.
__device__ inline v16h load_frag_h(const _Float16* __restrict__ p, int ld, int row0,
                                   int k0, int ln, int half) {
  const _Float16* base = p + (row0 + ln) * ld + k0 + (half << 3);
  v8h lo = *(const v8h*)base;
  v8h hi = *(const v8h*)(base + 16);
  v16h f;
#pragma unroll
  for (int i = 0; i < 8; ++i) { f[i] = lo[i]; f[8 + i] = hi[i]; }
  return f;
}

// ---------------- prep: fold BN into weights ----------------
__global__ void prep_kernel(const float* __restrict__ w, const float* __restrict__ bw,
                            const float* __restrict__ g, const float* __restrict__ beta,
                            const float* __restrict__ mean, const float* __restrict__ var,
                            float extra, _Float16* __restrict__ wh, float* __restrict__ bout) {
  int o = blockIdx.x, c = threadIdx.x;
  float scale = extra;
  if (g) scale *= g[o] * rsqrtf(var[o] + 1e-5f);
  wh[o * CC + c] = (_Float16)(w[o * CC + c] * scale);
  if (c == 0 && bout) {
    float s2 = g[o] * rsqrtf(var[o] + 1e-5f);
    bout[o] = ((bw[o] - mean[o]) * s2 + beta[o]) * extra;
  }
}

// ---------------- projection GEMM: P = W' * (x + pe) + bias' ----------------
// Each wave: 64 output rows x 16 columns.  mode 0 -> store [b,h,l,d]; mode 1 -> [b,h,d,l].
__global__ __launch_bounds__(256) void proj_kernel(
    const float* __restrict__ x, const float* __restrict__ pe,
    const _Float16* __restrict__ wh, const float* __restrict__ bias,
    _Float16* __restrict__ outp, int mode) {
  int wave = blockIdx.x * (blockDim.x >> 5) + (threadIdx.x >> 5);
  int lane = threadIdx.x & 31;
  int ln = lane & 15, half = lane >> 4;
  int ct = wave >> 2;  // column tile 0..511
  int rg = wave & 3;   // row group 0..3 (64 rows each)
  int col = ct * 16 + ln;
  int b = col >> 11, l = col & (LSEQ - 1);

  v8f acc[4] = {};
#pragma unroll
  for (int kc = 0; kc < 8; ++kc) {
    int kb = kc * 32 + (half << 3);
    // B fragment: X[c, col] = x[b][c][l] + pe[c][l]  (strided fp32 gather -> f16)
    v16h bf;
#pragma unroll
    for (int j = 0; j < 8; ++j) {
      int c0 = kb + j, c1 = kb + 16 + j;
      bf[j]     = (_Float16)(x[(b * CC + c0) * LSEQ + l] + pe[c0 * LSEQ + l]);
      bf[8 + j] = (_Float16)(x[(b * CC + c1) * LSEQ + l] + pe[c1 * LSEQ + l]);
    }
#pragma unroll
    for (int mt = 0; mt < 4; ++mt) {
      v16h af = load_frag_h(wh, CC, rg * 64 + mt * 16, kc * 32, ln, half);
      acc[mt] = WMMA_F16(af, bf, acc[mt]);
    }
  }
#pragma unroll
  for (int mt = 0; mt < 4; ++mt) {
#pragma unroll
    for (int r = 0; r < 8; ++r) {
      int o = rg * 64 + mt * 16 + r + half * 8;  // C/D layout: row = r + 8*half
      float val = acc[mt][r] + bias[o];
      int h = o >> 5, d = o & 31;
      int idx = (mode == 0) ? ((b * NH + h) * LSEQ + l) * DD + d
                            : ((b * NH + h) * DD + d) * LSEQ + l;
      outp[idx] = (_Float16)val;
    }
  }
}

// ---------------- attention: flash-style, one 16-query tile per wave ----------------
__global__ __launch_bounds__(128) void attn_kernel(
    const _Float16* __restrict__ qp, const _Float16* __restrict__ kp,
    const _Float16* __restrict__ vp, _Float16* __restrict__ att) {
  __shared__ __align__(16) _Float16 lds[4][16 * 64];  // per-wave 16x64 P tile
  int wslot = threadIdx.x >> 5;
  int wave = blockIdx.x * 4 + wslot;
  int lane = threadIdx.x & 31;
  int ln = lane & 15, half = lane >> 4;
  int ltile = wave & 127;
  int h = (wave >> 7) & 7;
  int b = wave >> 10;

  const _Float16* qbase = qp + (size_t)(b * NH + h) * LSEQ * DD;
  const _Float16* kbase = kp + (size_t)(b * NH + h) * LSEQ * DD;
  const _Float16* vbase = vp + (size_t)(b * NH + h) * DD * LSEQ;
  _Float16* lbuf = lds[wslot];

  // Q fragment: D=32 == one WMMA K step, loaded once (scale 1/sqrt(D) pre-folded).
  v16h aq = load_frag_h(qbase, DD, ltile * 16, 0, ln, half);

  v8f o0 = {}, o1 = {};
  float mrun[8], srun[8];
#pragma unroll
  for (int r = 0; r < 8; ++r) { mrun[r] = -1e30f; srun[r] = 0.0f; }

  for (int it = 0; it < LSEQ / 64; ++it) {
    int mbase = it * 64;
    v8f s[4];
#pragma unroll
    for (int t = 0; t < 4; ++t) {
      v16h bk = load_frag_h(kbase, DD, mbase + t * 16, 0, ln, half);
      v8f z = {};
      s[t] = WMMA_F16(aq, bk, z);
    }
    // online softmax over this 64-key slab; row = r + 8*half, row data across 16 lanes
#pragma unroll
    for (int r = 0; r < 8; ++r) {
      float mx = fmaxf(fmaxf(s[0][r], s[1][r]), fmaxf(s[2][r], s[3][r]));
#pragma unroll
      for (int off = 1; off < 16; off <<= 1) mx = fmaxf(mx, __shfl_xor(mx, off, 32));
      float mnew = fmaxf(mrun[r], mx);
      float resc = __expf(mrun[r] - mnew);
      float ls = 0.0f;
#pragma unroll
      for (int t = 0; t < 4; ++t) {
        float p = __expf(s[t][r] - mnew);
        s[t][r] = p;
        ls += p;
      }
#pragma unroll
      for (int off = 1; off < 16; off <<= 1) ls += __shfl_xor(ls, off, 32);
      srun[r] = srun[r] * resc + ls;
      mrun[r] = mnew;
      o0[r] *= resc;
      o1[r] *= resc;
    }
    // transpose P (C/D layout -> A layout) through LDS as f16
#pragma unroll
    for (int t = 0; t < 4; ++t)
#pragma unroll
      for (int r = 0; r < 8; ++r)
        lbuf[(r + half * 8) * 64 + t * 16 + ln] = (_Float16)s[t][r];
    __syncthreads();
#pragma unroll
    for (int mc = 0; mc < 2; ++mc) {
      v16h ap = load_frag_h(lbuf, 64, 0, mc * 32, ln, half);
      v16h bv0 = load_frag_h(vbase, LSEQ, 0, mbase + mc * 32, ln, half);
      o0 = WMMA_F16(ap, bv0, o0);
      v16h bv1 = load_frag_h(vbase, LSEQ, 16, mbase + mc * 32, ln, half);
      o1 = WMMA_F16(ap, bv1, o1);
    }
    __syncthreads();
  }
  // normalize and store [b, l, c] f16 (l-major so out-proj B fragments are contiguous)
#pragma unroll
  for (int r = 0; r < 8; ++r) {
    float inv = 1.0f / srun[r];
    int l = ltile * 16 + r + half * 8;
    size_t rowoff = ((size_t)(b * LSEQ + l)) * CC + h * 32;
    att[rowoff + ln]      = (_Float16)(o0[r] * inv);
    att[rowoff + 16 + ln] = (_Float16)(o1[r] * inv);
  }
}

// ---------------- output projection + bias + residual ----------------
__global__ __launch_bounds__(256) void outproj_kernel(
    const _Float16* __restrict__ att, const _Float16* __restrict__ woh,
    const float* __restrict__ bo, const float* __restrict__ qin,
    float* __restrict__ out) {
  int wave = blockIdx.x * (blockDim.x >> 5) + (threadIdx.x >> 5);
  int lane = threadIdx.x & 31;
  int ln = lane & 15, half = lane >> 4;
  int ct = wave >> 2;
  int rg = wave & 3;
  int col = ct * 16 + ln;
  int b = col >> 11, l = col & (LSEQ - 1);

  v8f acc[4] = {};
#pragma unroll
  for (int kc = 0; kc < 8; ++kc) {
    // B fragment: att rows are contiguous f16
    v16h bf = load_frag_h(att, CC, col - ln, kc * 32, ln, half);  // row0+ln == col
#pragma unroll
    for (int mt = 0; mt < 4; ++mt) {
      v16h af = load_frag_h(woh, CC, rg * 64 + mt * 16, kc * 32, ln, half);
      acc[mt] = WMMA_F16(af, bf, acc[mt]);
    }
  }
#pragma unroll
  for (int mt = 0; mt < 4; ++mt) {
#pragma unroll
    for (int r = 0; r < 8; ++r) {
      int o = rg * 64 + mt * 16 + r + half * 8;
      int idx = (b * CC + o) * LSEQ + l;
      out[idx] = acc[mt][r] + bo[o] + qin[idx];
    }
  }
}

extern "C" void kernel_launch(void* const* d_in, const int* in_sizes, int n_in,
                              void* d_out, int out_size, void* d_ws, size_t ws_size,
                              hipStream_t stream) {
  (void)in_sizes; (void)n_in; (void)out_size; (void)ws_size;
  const float* v     = (const float*)d_in[0];
  const float* k     = (const float*)d_in[1];
  const float* q     = (const float*)d_in[2];
  const float* pe_q  = (const float*)d_in[3];
  const float* pe_vk = (const float*)d_in[4];
  const float* wq = (const float*)d_in[5];  const float* bq = (const float*)d_in[6];
  const float* gq = (const float*)d_in[7];  const float* betaq = (const float*)d_in[8];
  const float* mq = (const float*)d_in[9];  const float* varq = (const float*)d_in[10];
  const float* wk = (const float*)d_in[11]; const float* bk = (const float*)d_in[12];
  const float* gk = (const float*)d_in[13]; const float* betak = (const float*)d_in[14];
  const float* mk = (const float*)d_in[15]; const float* vark = (const float*)d_in[16];
  const float* wv = (const float*)d_in[17]; const float* bv = (const float*)d_in[18];
  const float* gv = (const float*)d_in[19]; const float* betav = (const float*)d_in[20];
  const float* mv = (const float*)d_in[21]; const float* varv = (const float*)d_in[22];
  const float* wo = (const float*)d_in[23]; const float* bo = (const float*)d_in[24];

  char* ws = (char*)d_ws;
  _Float16* wqh = (_Float16*)(ws + WS_WQH);
  _Float16* wkh = (_Float16*)(ws + WS_WKH);
  _Float16* wvh = (_Float16*)(ws + WS_WVH);
  _Float16* woh = (_Float16*)(ws + WS_WOH);
  float* biasq = (float*)(ws + WS_BQ);
  float* biask = (float*)(ws + WS_BK);
  float* biasv = (float*)(ws + WS_BV);
  _Float16* qpb = (_Float16*)(ws + WS_QP);
  _Float16* kpb = (_Float16*)(ws + WS_KP);
  _Float16* vpb = (_Float16*)(ws + WS_VP);
  _Float16* attb = (_Float16*)(ws + WS_ATT);

  const float invSqrtD = 0.17677669529663687f;  // 1/sqrt(32), folded into Q path

  prep_kernel<<<CC, CC, 0, stream>>>(wq, bq, gq, betaq, mq, varq, invSqrtD, wqh, biasq);
  prep_kernel<<<CC, CC, 0, stream>>>(wk, bk, gk, betak, mk, vark, 1.0f, wkh, biask);
  prep_kernel<<<CC, CC, 0, stream>>>(wv, bv, gv, betav, mv, varv, 1.0f, wvh, biasv);
  prep_kernel<<<CC, CC, 0, stream>>>(wo, bo, nullptr, nullptr, nullptr, nullptr, 1.0f, woh, nullptr);

  // 2048 waves per GEMM: 512 col-tiles x 4 row-groups, 8 waves/block
  proj_kernel<<<256, 256, 0, stream>>>(q, pe_q,  wqh, biasq, qpb, 0);
  proj_kernel<<<256, 256, 0, stream>>>(k, pe_vk, wkh, biask, kpb, 0);
  proj_kernel<<<256, 256, 0, stream>>>(v, pe_vk, wvh, biasv, vpb, 1);

  // 4096 waves: one 16-query tile per wave, 4 waves/block
  attn_kernel<<<1024, 128, 0, stream>>>(qpb, kpb, vpb, attb);

  outproj_kernel<<<256, 256, 0, stream>>>(attb, woh, bo, q, (float*)d_out);
}
